// CRIS_44367012167708
// MI455X (gfx1250) — matrix operand order repaired and checked by reading, not compiled, not run on hardware
//
#include <hip/hip_runtime.h>
#include <hip/hip_bf16.h>

typedef __attribute__((ext_vector_type(16))) _Float16 v16h;
typedef __attribute__((ext_vector_type(8)))  _Float16 v8h;
typedef __attribute__((ext_vector_type(8)))  float    v8f;

namespace cris {

constexpr int Bx    = 16;
constexpr int Dm    = 768;
constexpr int NHd   = 12;
constexpr int HDd   = 64;
constexpr int NL    = 12;
constexpr int NCc   = 27;
constexpr int MLPd  = 3072;
constexpr int PROJd = 512;
constexpr int IMGd  = 224;
constexpr int GR    = 14;          // 224/16
constexpr int HWd   = 196;         // 14*14
constexpr int Sd    = NCc + HWd;   // 223
constexpr int SP    = 288;         // padded sequence (multiple of 32)
constexpr int Mrows = Bx * SP;     // 4608 (multiple of 128)
constexpr int QKVN  = 3 * Dm;      // 2304
constexpr int MP    = 3200;        // padded patch-rows (25*128)
constexpr float NEGV  = -1e9f;
constexpr float SCALE = 0.125f;    // 1/sqrt(64)

// ---------------- WMMA GEMM ----------------
constexpr int BM = 128, BN = 128, BK = 32;
constexpr int LDA_S = BK + 8;   // 40 halves
constexpr int LDB_S = BN + 8;   // 136 halves

__device__ inline v16h load_frag_a(const _Float16* sp, int lane) {
  // A 16x32 f16: lanes 0-15 -> M=lane, K in {0..7,16..23}; lanes 16-31 -> K in {8..15,24..31}
  const int m  = lane & 15;
  const int kb = (lane & 16) ? 8 : 0;
  union { v16h v; unsigned u[8]; } r;
#pragma unroll
  for (int i = 0; i < 8; ++i) {
    const int k = (i < 4) ? (kb + 2 * i) : (16 + kb + 2 * (i - 4));
    r.u[i] = *(const unsigned*)(sp + m * LDA_S + k);
  }
  return r.v;
}

__device__ inline v16h load_frag_b(const _Float16* sp, int lane) {
  // B 32x16 f16: lanes 0-15 -> N=lane, K in {0..7,16..23}; lanes 16-31 -> K in {8..15,24..31}
  const int n  = lane & 15;
  const int kb = (lane & 16) ? 8 : 0;
  union { v16h v; _Float16 e[16]; } r;
#pragma unroll
  for (int s = 0; s < 16; ++s) {
    const int k = (s < 8) ? (kb + s) : (16 + kb + (s - 8));
    r.e[s] = sp[k * LDB_S + n];
  }
  return r.v;
}

// C = act(A(MxK,f32) * B(KxN,f32) + bias) [+ Res]; all dims multiples of 128/128/32.
// Double-buffered LDS ping-pong with register staging: global loads for tile t+1
// are issued before the 8 WMMAs of tile t; the dependent cvt+ds_store (and its
// s_wait_loadcnt) lands after the WMMAs, hiding HBM latency behind matrix math.
template <bool TRANSB, bool HAS_BIAS, bool GELU_ACT, bool HAS_RES>
__global__ __launch_bounds__(256)
void gemm_wmma(const float* __restrict__ A, const float* __restrict__ Bw,
               const float* __restrict__ bias, const float* Res,
               float* C, int Mdim, int Ndim, int Kdim) {
  (void)Mdim;
  __shared__ __align__(16) _Float16 As[2][BM * LDA_S];
  __shared__ __align__(16) _Float16 Bs[2][BK * LDB_S];

  const int tid  = threadIdx.x;
  const int lane = tid & 31;
  const int w    = tid >> 5;
  const int wm   = w & 1;    // 2 waves along M (64 rows each)
  const int wn   = w >> 1;   // 4 waves along N (32 cols each)
  const int bm   = blockIdx.y * BM;
  const int bn   = blockIdx.x * BN;

  v8f zero = {};
  v8f acc[4][2];
#pragma unroll
  for (int mi = 0; mi < 4; ++mi)
#pragma unroll
    for (int ni = 0; ni < 2; ++ni) acc[mi][ni] = zero;

  float4 av[4], bv[4];

  // issue all 8 global_load_b128 for K-tile kt into registers
  auto load_global = [&](int kt) {
#pragma unroll
    for (int i = 0; i < 4; ++i) {
      const int id = tid + i * 256;
      av[i] = *(const float4*)(A + (size_t)(bm + (id >> 3)) * Kdim + kt + ((id & 7) << 2));
    }
    if (!TRANSB) {
#pragma unroll
      for (int i = 0; i < 4; ++i) {
        const int id = tid + i * 256;
        bv[i] = *(const float4*)(Bw + (size_t)(kt + (id >> 5)) * Ndim + bn + ((id & 31) << 2));
      }
    } else {
#pragma unroll
      for (int i = 0; i < 4; ++i) {
        const int id = tid + i * 256;
        bv[i] = *(const float4*)(Bw + (size_t)(bn + (id >> 3)) * Kdim + kt + ((id & 7) << 2));
      }
    }
  };

  // convert f32 -> f16 and store staged registers to LDS buffer `buf`
  auto store_lds = [&](int buf) {
#pragma unroll
    for (int i = 0; i < 4; ++i) {
      const int id = tid + i * 256;
      _Float16* dst = &As[buf][(id >> 3) * LDA_S + ((id & 7) << 2)];
      dst[0] = (_Float16)av[i].x; dst[1] = (_Float16)av[i].y;
      dst[2] = (_Float16)av[i].z; dst[3] = (_Float16)av[i].w;
    }
    if (!TRANSB) {
#pragma unroll
      for (int i = 0; i < 4; ++i) {
        const int id = tid + i * 256;
        _Float16* dst = &Bs[buf][(id >> 5) * LDB_S + ((id & 31) << 2)];
        dst[0] = (_Float16)bv[i].x; dst[1] = (_Float16)bv[i].y;
        dst[2] = (_Float16)bv[i].z; dst[3] = (_Float16)bv[i].w;
      }
    } else {
#pragma unroll
      for (int i = 0; i < 4; ++i) {
        const int id = tid + i * 256;
        const int n = id >> 3, k4 = (id & 7) << 2;
        Bs[buf][(k4 + 0) * LDB_S + n] = (_Float16)bv[i].x;
        Bs[buf][(k4 + 1) * LDB_S + n] = (_Float16)bv[i].y;
        Bs[buf][(k4 + 2) * LDB_S + n] = (_Float16)bv[i].z;
        Bs[buf][(k4 + 3) * LDB_S + n] = (_Float16)bv[i].w;
      }
    }
  };

  const int nk = Kdim / BK;
  load_global(0);
  store_lds(0);
  __syncthreads();

  for (int t = 0; t < nk; ++t) {
    const int cur = t & 1;
    if (t + 1 < nk) {
      load_global((t + 1) * BK);            // loads in flight during WMMAs below
      if (t + 2 < nk)                       // speculative prefetch 2 tiles ahead
        __builtin_prefetch(A + (size_t)(bm + (tid >> 1)) * Kdim + (t + 2) * BK, 0, 0);
    }

    v16h afr[4], bfr[2];
#pragma unroll
    for (int mi = 0; mi < 4; ++mi)
      afr[mi] = load_frag_a(&As[cur][(wm * 64 + mi * 16) * LDA_S], lane);
#pragma unroll
    for (int ni = 0; ni < 2; ++ni)
      bfr[ni] = load_frag_b(&Bs[cur][wn * 32 + ni * 16], lane);

#pragma unroll
    for (int mi = 0; mi < 4; ++mi)
#pragma unroll
      for (int ni = 0; ni < 2; ++ni)
        acc[mi][ni] = __builtin_amdgcn_wmma_f32_16x16x32_f16(
            false, afr[mi], false, bfr[ni], (short)0, acc[mi][ni], false, false);

    if (t + 1 < nk) {
      __syncthreads();          // previous readers of buffer 1-cur are done
      store_lds(1 - cur);       // s_wait_loadcnt lands here, after the WMMAs
      __syncthreads();
    }
  }

  // ---- epilogue
  const int mo = ((lane >> 4) & 1) * 8;
  const int nl = lane & 15;
#pragma unroll
  for (int mi = 0; mi < 4; ++mi) {
#pragma unroll
    for (int ni = 0; ni < 2; ++ni) {
      const int ncc  = bn + wn * 32 + ni * 16 + nl;
      const float bv2 = HAS_BIAS ? bias[ncc] : 0.0f;
#pragma unroll
      for (int r = 0; r < 8; ++r) {
        const int m = bm + wm * 64 + mi * 16 + mo + r;
        float v = acc[mi][ni][r] + bv2;
        if (GELU_ACT) v = v / (1.0f + __expf(-1.702f * v));   // quick-GELU
        if (HAS_RES)  v += Res[(size_t)m * Ndim + ncc];
        C[(size_t)m * Ndim + ncc] = v;
      }
    }
  }
}

// ---------------- im2col (patch conv -> GEMM A) ----------------
__global__ __launch_bounds__(256)
void im2col_kernel(const float* __restrict__ img, float* __restrict__ P) {
  const int idx = blockIdx.x * 256 + threadIdx.x;
  if (idx >= MP * Dm) return;
  const int row = idx / Dm, k = idx % Dm;
  if (row >= Bx * HWd) { P[idx] = 0.0f; return; }
  const int b = row / HWd, p = row % HWd;
  const int gy = p / GR, gx = p % GR;
  const int c = k >> 8, rem = k & 255, py = rem >> 4, px = rem & 15;
  P[idx] = img[((size_t)(b * 3 + c)) * IMGd * IMGd + (gy * 16 + py) * IMGd + (gx * 16 + px)];
}

// ---------------- assemble tokens + pos embed ----------------
__global__ __launch_bounds__(256)
void assemble_x(const float* __restrict__ Pout, const float* __restrict__ cls,
                const float* __restrict__ pb, const float* __restrict__ pos,
                float* __restrict__ X) {
  const int idx = blockIdx.x * 256 + threadIdx.x;
  if (idx >= Mrows * Dm) return;
  const int row = idx / Dm, d = idx % Dm;
  const int b = row / SP, s = row % SP;
  float v;
  if (s < NCc)      v = cls[s * Dm + d] + pos[(size_t)s * Dm + d];
  else if (s < Sd)  v = Pout[((size_t)(b * HWd + (s - NCc))) * Dm + d] + pb[d] + pos[(size_t)s * Dm + d];
  else              v = 0.0f;
  X[idx] = v;
}

// ---------------- LayerNorm (row-wise) ----------------
__global__ __launch_bounds__(256)
void layernorm_kernel(const float* __restrict__ X, const float* __restrict__ w,
                      const float* __restrict__ bb, float* __restrict__ Out) {
  const int row = blockIdx.x;
  const int s   = row % SP;
  const int tid = threadIdx.x;
  const float* x = X + (size_t)row * Dm;
  float sum = 0.0f, sq = 0.0f;
  for (int d = tid; d < Dm; d += 256) { const float v = x[d]; sum += v; sq += v * v; }
  __shared__ float r1[256], r2[256];
  r1[tid] = sum; r2[tid] = sq;
  __syncthreads();
  for (int o = 128; o > 0; o >>= 1) {
    if (tid < o) { r1[tid] += r1[tid + o]; r2[tid] += r2[tid + o]; }
    __syncthreads();
  }
  const float mean = r1[0] * (1.0f / Dm);
  const float var  = r2[0] * (1.0f / Dm) - mean * mean;
  const float rs   = rsqrtf(var + 1e-5f);
  float* o = Out + (size_t)row * Dm;
  if (s >= Sd) { for (int d = tid; d < Dm; d += 256) o[d] = 0.0f; }
  else { for (int d = tid; d < Dm; d += 256) o[d] = (x[d] - mean) * rs * w[d] + bb[d]; }
}

// ---------------- masked flash attention (per b,h) ----------------
__global__ __launch_bounds__(256)
void attention_kernel(const float* __restrict__ QKV, const int* __restrict__ aom,
                      float* __restrict__ O) {
  extern __shared__ char smem[];
  _Float16* Ksh = (_Float16*)smem;          // Sd*HDd
  _Float16* Vsh = Ksh + Sd * HDd;           // Sd*HDd
  int* aomsh    = (int*)(Vsh + Sd * HDd);   // HWd
  const int bh = blockIdx.x;
  const int b = bh / NHd, h = bh % NHd;
  const int tid = threadIdx.x;

  for (int idx = tid; idx < Sd * HDd; idx += 256) {
    const int s = idx / HDd, d = idx % HDd;
    const size_t base = ((size_t)(b * SP + s)) * QKVN;
    Ksh[idx] = (_Float16)QKV[base + Dm + h * HDd + d];
    Vsh[idx] = (_Float16)QKV[base + 2 * Dm + h * HDd + d];
  }
  for (int idx = tid; idx < HWd; idx += 256) aomsh[idx] = aom[b * HWd + idx];
  __syncthreads();

  for (int q = tid; q < Sd; q += 256) {
    float qv[HDd];
    const size_t qbase = ((size_t)(b * SP + q)) * QKVN + h * HDd;
#pragma unroll
    for (int d = 0; d < HDd; ++d) qv[d] = QKV[qbase + d] * SCALE;
    float m = -1e30f, l = 0.0f;
    float acc[HDd];
#pragma unroll
    for (int d = 0; d < HDd; ++d) acc[d] = 0.0f;
    for (int k = 0; k < Sd; ++k) {
      float dot = 0.0f;
#pragma unroll
      for (int d = 0; d < HDd; ++d) dot += qv[d] * (float)Ksh[k * HDd + d];
      float msk;
      if (q < NCc) msk = (k < NCc) ? ((q == k) ? 0.0f : NEGV)
                                   : ((aomsh[k - NCc] == q) ? 0.0f : NEGV);
      else         msk = (k < NCc) ? NEGV : 0.0f;
      const float sc = dot + msk;
      const float mn = fmaxf(m, sc);
      const float corr = __expf(m - mn);
      const float p = __expf(sc - mn);
      l = l * corr + p;
#pragma unroll
      for (int d = 0; d < HDd; ++d) acc[d] = acc[d] * corr + p * (float)Vsh[k * HDd + d];
      m = mn;
    }
    const float inv = 1.0f / l;
    const size_t obase = ((size_t)(b * SP + q)) * Dm + h * HDd;
#pragma unroll
    for (int d = 0; d < HDd; ++d) O[obase + d] = acc[d] * inv;
  }
  // zero padded rows (so out-proj residual stays clean)
  for (int idx = tid; idx < (SP - Sd) * HDd; idx += 256) {
    const int r = Sd + idx / HDd, d = idx % HDd;
    O[((size_t)(b * SP + r)) * Dm + h * HDd + d] = 0.0f;
  }
}

// ---------------- cosine scoring ----------------
__global__ __launch_bounds__(256)
void score_kernel(const float* __restrict__ T, const float* __restrict__ sents,
                  float* __restrict__ predId) {
  const int b = blockIdx.x / NCc, n = blockIdx.x % NCc;
  const int tid = threadIdx.x;
  const float* t  = T + ((size_t)(b * SP + n)) * PROJd;
  const float* sv = sents + (size_t)b * PROJd;
  float dp = 0.0f, tn = 0.0f, sn = 0.0f;
  for (int j = tid; j < PROJd; j += 256) {
    const float tv = t[j], s = sv[j];
    dp += tv * s; tn += tv * tv; sn += s * s;
  }
  __shared__ float r0[256], r1[256], r2[256];
  r0[tid] = dp; r1[tid] = tn; r2[tid] = sn;
  __syncthreads();
  for (int o = 128; o > 0; o >>= 1) {
    if (tid < o) { r0[tid] += r0[tid + o]; r1[tid] += r1[tid + o]; r2[tid] += r2[tid + o]; }
    __syncthreads();
  }
  if (tid == 0) predId[blockIdx.x] = r0[0] * rsqrtf(r1[0]) * rsqrtf(r2[0]);
}

__global__ __launch_bounds__(256)
void gather_kernel(const float* __restrict__ predId, const int* __restrict__ om,
                   float* __restrict__ out, int n) {
  const int idx = blockIdx.x * 256 + threadIdx.x;
  if (idx >= n) return;
  const int b = idx / (IMGd * IMGd);
  out[idx] = predId[b * NCc + om[idx]];
}

} // namespace cris

extern "C" void kernel_launch(void* const* d_in, const int* in_sizes, int n_in,
                              void* d_out, int out_size, void* d_ws, size_t ws_size,
                              hipStream_t stream) {
  using namespace cris;
  (void)in_sizes; (void)n_in; (void)out_size; (void)ws_size;

  const float* img     = (const float*)d_in[0];
  const float* sents   = (const float*)d_in[1];
  const int*   aom     = (const int*)d_in[2];
  const int*   om      = (const int*)d_in[3];
  const float* cls     = (const float*)d_in[4];
  const float* patch_w = (const float*)d_in[5];
  const float* patch_b = (const float*)d_in[6];
  const float* pos     = (const float*)d_in[7];
  const float* ln1w    = (const float*)d_in[8];
  const float* ln1b    = (const float*)d_in[9];
  const float* qkvw    = (const float*)d_in[10];
  const float* qkvb    = (const float*)d_in[11];
  const float* outw    = (const float*)d_in[12];
  const float* outb    = (const float*)d_in[13];
  const float* ln2w    = (const float*)d_in[14];
  const float* ln2b    = (const float*)d_in[15];
  const float* w1      = (const float*)d_in[16];
  const float* b1      = (const float*)d_in[17];
  const float* w2      = (const float*)d_in[18];
  const float* b2      = (const float*)d_in[19];
  const float* lpw     = (const float*)d_in[20];
  const float* lpb     = (const float*)d_in[21];
  const float* proj    = (const float*)d_in[22];
  float* out = (float*)d_out;

  // workspace layout (fp32)
  const size_t MX = (size_t)Mrows * Dm;      // 3,538,944
  const size_t MQ = (size_t)Mrows * QKVN;    // 10,616,832
  float* ws   = (float*)d_ws;
  float* X    = ws;
  float* Hb   = X + MX;
  float* QKVb = Hb + MX;
  float* Ob   = QKVb + MQ;
  float* Gb   = Ob + MX;
  float* P    = Gb;     // alias (im2col rows, MP*Dm < Mrows*MLPd)
  float* Pout = QKVb;   // alias (conv out, MP*Dm < MQ)
  float* T    = Ob;     // alias (projected tokens, Mrows*PROJd < MX)
  float* predId = Hb;   // alias (B*NC floats)

  const int ATT_SMEM = Sd * HDd * 2 * (int)sizeof(_Float16) + HWd * (int)sizeof(int);
  hipFuncSetAttribute((const void*)attention_kernel,
                      hipFuncAttributeMaxDynamicSharedMemorySize, ATT_SMEM);

  // ---- patch embedding: im2col + GEMM (patch_w is NxK -> TRANSB)
  im2col_kernel<<<(MP * Dm + 255) / 256, 256, 0, stream>>>(img, P);
  gemm_wmma<true, false, false, false>
      <<<dim3(Dm / BN, MP / BM), 256, 0, stream>>>(P, patch_w, nullptr, nullptr,
                                                   Pout, MP, Dm, Dm);
  assemble_x<<<((int)MX + 255) / 256, 256, 0, stream>>>(Pout, cls, patch_b, pos, X);

  // ---- transformer layers
  for (int l = 0; l < NL; ++l) {
    layernorm_kernel<<<Mrows, 256, 0, stream>>>(X, ln1w + l * Dm, ln1b + l * Dm, Hb);
    gemm_wmma<false, true, false, false>
        <<<dim3(QKVN / BN, Mrows / BM), 256, 0, stream>>>(
            Hb, qkvw + (size_t)l * Dm * QKVN, qkvb + (size_t)l * QKVN, nullptr,
            QKVb, Mrows, QKVN, Dm);
    attention_kernel<<<Bx * NHd, 256, ATT_SMEM, stream>>>(QKVb, aom, Ob);
    gemm_wmma<false, true, false, true>
        <<<dim3(Dm / BN, Mrows / BM), 256, 0, stream>>>(
            Ob, outw + (size_t)l * Dm * Dm, outb + (size_t)l * Dm, X,
            X, Mrows, Dm, Dm);
    layernorm_kernel<<<Mrows, 256, 0, stream>>>(X, ln2w + l * Dm, ln2b + l * Dm, Hb);
    gemm_wmma<false, true, true, false>
        <<<dim3(MLPd / BN, Mrows / BM), 256, 0, stream>>>(
            Hb, w1 + (size_t)l * Dm * MLPd, b1 + (size_t)l * MLPd, nullptr,
            Gb, Mrows, MLPd, Dm);
    gemm_wmma<false, true, false, true>
        <<<dim3(Dm / BN, Mrows / BM), 256, 0, stream>>>(
            Gb, w2 + (size_t)l * MLPd * Dm, b2 + (size_t)l * Dm, X,
            X, Mrows, Dm, MLPd);
  }

  // ---- final LN + projection + scoring + gather
  layernorm_kernel<<<Mrows, 256, 0, stream>>>(X, lpw, lpb, Hb);
  gemm_wmma<false, false, false, false>
      <<<dim3(PROJd / BN, Mrows / BM), 256, 0, stream>>>(Hb, proj, nullptr, nullptr,
                                                         T, Mrows, PROJd, Dm);
  score_kernel<<<Bx * NCc, 256, 0, stream>>>(T, sents, predId);
  const int nOut = Bx * IMGd * IMGd;
  gather_kernel<<<(nOut + 255) / 256, 256, 0, stream>>>(predId, om, out, nOut);
}
